// PointCraftPlusPlusModel_16475494548054
// MI455X (gfx1250) — matrix operand order repaired and checked by reading, not compiled, not run on hardware
//
#include <hip/hip_runtime.h>
#include <hip/hip_bf16.h>

typedef __attribute__((ext_vector_type(16))) _Float16 v16h;
typedef __attribute__((ext_vector_type(8)))  _Float16 v8h;
typedef __attribute__((ext_vector_type(8)))  float    v8f;
typedef __attribute__((ext_vector_type(4)))  float    v4f;

#define WMMA_F32_F16(a, b, c) \
  __builtin_amdgcn_wmma_f32_16x16x32_f16(false, (a), false, (b), (short)0, (c), false, false)

// ---------------------------------------------------------------------------
// CDNA5 wave32 WMMA fragment layouts (CDNA5 ISA §7.12.2):
// A (16x32 f16): lane row M=lane&15; K = 8*(lane>>4) + (j<8 ? j : j+8)
//   -> this code loads A as K = 16*(lane>>4) + j instead (16 contiguous, same as B)
// B (32x16 f16): lane col N=lane&15; code uses K = 16*(lane>>4) + j  -> 16 contiguous, as for A
// C/D (16x16 f32): M = r + 8*(lane>>4), N = lane&15
// ---------------------------------------------------------------------------
static __device__ __forceinline__ v16h frag_from2(v8h lo, v8h hi) {
  v16h r;
#pragma unroll
  for (int j = 0; j < 8; ++j) { r[j] = lo[j]; r[j + 8] = hi[j]; }
  return r;
}

// A fragment from f16 row-major buffer (16B-aligned rows).
static __device__ __forceinline__ v16h load_a_f16(const _Float16* __restrict__ p,
                                                  int stride, int lane) {
  int m = lane & 15, kb = (lane >> 4) * 16;
  const _Float16* row = p + m * stride + kb;
  return frag_from2(*(const v8h*)row, *(const v8h*)(row + 8));
}

// B fragment from 16 contiguous f16 (16B-aligned).
static __device__ __forceinline__ v16h load16_f16(const _Float16* __restrict__ p) {
  return frag_from2(*(const v8h*)p, *(const v8h*)(p + 8));
}

// A fragment from f32 row-major buffer, cvt to f16 (vectorized b128 loads).
static __device__ __forceinline__ v16h load_a_f32(const float* __restrict__ p,
                                                  int stride, int lane) {
  int m = lane & 15, kb = (lane >> 4) * 16;
  const float* row = p + m * stride + kb;
  v4f c0 = *(const v4f*)(row);
  v4f c1 = *(const v4f*)(row + 4);
  v4f c2 = *(const v4f*)(row + 8);
  v4f c3 = *(const v4f*)(row + 12);
  v16h a;
#pragma unroll
  for (int j = 0; j < 4; ++j) {
    a[j]      = (_Float16)c0[j];
    a[4 + j]  = (_Float16)c1[j];
    a[8 + j]  = (_Float16)c2[j];
    a[12 + j] = (_Float16)c3[j];
  }
  return a;
}

// B fragment: B[k][n] = W[n][k], W row-major [N][K] f32; p -> W(n0,k0).
static __device__ __forceinline__ v16h load_bT_f32(const float* __restrict__ p,
                                                   int stride, int lane) {
  int n = lane & 15, kb = (lane >> 4) * 16;
  const float* row = p + n * stride + kb;
  v4f c0 = *(const v4f*)(row);
  v4f c1 = *(const v4f*)(row + 4);
  v4f c2 = *(const v4f*)(row + 8);
  v4f c3 = *(const v4f*)(row + 12);
  v16h b;
#pragma unroll
  for (int j = 0; j < 4; ++j) {
    b[j]      = (_Float16)c0[j];
    b[4 + j]  = (_Float16)c1[j];
    b[8 + j]  = (_Float16)c2[j];
    b[12 + j] = (_Float16)c3[j];
  }
  return b;
}

// ---------------------------------------------------------------------------
// Generic GEMM: C[M][N] = A[M][K] (f32) @ W[N][K]^T + bias[N]
// One wave -> 32x64 output block (2 M-tiles x 4 N-tiles): every W fragment
// feeds two WMMAs. M%32==0, N%64==0, K%32==0.
// ---------------------------------------------------------------------------
template <typename OutT>
__global__ __launch_bounds__(256) void gemm_bias_kernel(
    const float* __restrict__ A, const float* __restrict__ W,
    const float* __restrict__ bias, OutT* __restrict__ C,
    int M, int N, int K) {
  int lane = threadIdx.x & 31;
  int wave = (blockIdx.x * blockDim.x + threadIdx.x) >> 5;
  int nstrips = N >> 6;
  int mt = wave / nstrips;
  int ns = wave - mt * nstrips;
  if (mt * 32 >= M) return;
  int m0 = mt * 32, n0 = ns * 64;

  v8f acc[2][4] = {};
  for (int k0 = 0; k0 < K; k0 += 32) {
    v16h a0 = load_a_f32(A + m0 * K + k0, K, lane);
    v16h a1 = load_a_f32(A + (m0 + 16) * K + k0, K, lane);
#pragma unroll
    for (int t = 0; t < 4; ++t) {
      v16h b = load_bT_f32(W + (n0 + t * 16) * K + k0, K, lane);
      acc[0][t] = WMMA_F32_F16(a0, b, acc[0][t]);
      acc[1][t] = WMMA_F32_F16(a1, b, acc[1][t]);
    }
  }
  int rbase = (lane >> 4) * 8;
  int nc = lane & 15;
#pragma unroll
  for (int t = 0; t < 4; ++t) {
    int n = n0 + t * 16 + nc;
    float bv = bias[n];
#pragma unroll
    for (int half = 0; half < 2; ++half)
#pragma unroll
      for (int r = 0; r < 8; ++r) {
        int m = m0 + half * 16 + rbase + r;
        C[m * N + n] = (OutT)(acc[half][t][r] + bv);
      }
  }
}

// ---------------------------------------------------------------------------
// Fused attention, one wave per (group, head, 64-query tile). dh=32.
// K and V^T staged into LDS with zero padding to 64 keys; K/V fragments are
// hoisted into registers once and reused across all 4 query tiles.
// ---------------------------------------------------------------------------
#define KST 40   // K LDS row stride (halves): 80B, 16B-aligned
#define PST 72   // P and V^T LDS row stride (halves): 144B, 16B-aligned

__global__ __launch_bounds__(32) void attn_kernel(
    const _Float16* __restrict__ Qb, const _Float16* __restrict__ Kb,
    const _Float16* __restrict__ Vb, float* __restrict__ Ob,
    int kv_len, int qtiles,
    int q_row_stride, int kv_row_stride,
    int q_group_stride, int kv_group_stride, int o_group_stride,
    float scale) {
  __shared__ float    Ssm[64 * 65];     // scores (f32)
  __shared__ _Float16 Psm[64 * PST];    // softmax probs (f16)
  __shared__ _Float16 Ksm[64 * KST];    // K tile, zero-padded rows
  __shared__ _Float16 Vtsm[32 * PST];   // V^T tile (d-major), zero-padded cols

  int lane = threadIdx.x;
  int qt = blockIdx.x % qtiles;
  int h  = (blockIdx.x / qtiles) & 7;
  int g  = blockIdx.x / (qtiles * 8);
  const _Float16* Q  = Qb + g * q_group_stride + (qt * 64) * q_row_stride + h * 32;
  const _Float16* Kh = Kb + g * kv_group_stride + h * 32;
  const _Float16* Vh = Vb + g * kv_group_stride + h * 32;
  float* O = Ob + g * o_group_stride + (qt * 64) * 256 + h * 32;

  int lm = lane & 15, hi = lane >> 4;

  // -------- stage K (row-major, zero-pad rows >= kv_len) --------
  {
    v8h z = {};
#pragma unroll
    for (int rr = 0; rr < 2; ++rr) {
      int row = lane + rr * 32;
      _Float16* dst = &Ksm[row * KST];
      if (row < kv_len) {
        const _Float16* src = Kh + row * kv_row_stride;
        *(v8h*)(dst)      = *(const v8h*)(src);
        *(v8h*)(dst + 8)  = *(const v8h*)(src + 8);
        *(v8h*)(dst + 16) = *(const v8h*)(src + 16);
        *(v8h*)(dst + 24) = *(const v8h*)(src + 24);
      } else {
        *(v8h*)(dst) = z; *(v8h*)(dst + 8) = z;
        *(v8h*)(dst + 16) = z; *(v8h*)(dst + 24) = z;
      }
    }
  }
  // -------- stage V^T: Vtsm[d][k] = V[k][d] (coalesced reads) --------
  for (int kk = 0; kk < kv_len; ++kk)
    Vtsm[lane * PST + kk] = Vh[kk * kv_row_stride + lane];
  for (int kk = kv_len; kk < 64; ++kk)
    Vtsm[lane * PST + kk] = (_Float16)0.0f;

  // -------- S = scale * Q K^T (K fragments hoisted, reused 4x) --------
  {
    v16h kf[4];
#pragma unroll
    for (int nt = 0; nt < 4; ++nt)
      kf[nt] = load16_f16(&Ksm[(nt * 16 + lm) * KST + hi * 16]);
#pragma unroll
    for (int mt = 0; mt < 4; ++mt) {
      v16h qa = load_a_f16(Q + (mt * 16) * q_row_stride, q_row_stride, lane);
#pragma unroll
      for (int nt = 0; nt < 4; ++nt) {
        v8f acc = {};
        acc = WMMA_F32_F16(qa, kf[nt], acc);
#pragma unroll
        for (int r = 0; r < 8; ++r) {
          int q = mt * 16 + hi * 8 + r;
          int k = nt * 16 + lm;
          Ssm[q * 65 + k] = acc[r] * scale;   // padded cols never read
        }
      }
    }
  }

  // -------- softmax rows (2 per lane; in-wave LDS ops are ordered) --------
#pragma unroll
  for (int rr = 0; rr < 2; ++rr) {
    int q = lane + rr * 32;
    float mx = -1e30f;
    for (int k = 0; k < kv_len; ++k) mx = fmaxf(mx, Ssm[q * 65 + k]);
    float sum = 0.0f;
    for (int k = 0; k < kv_len; ++k) {
      float e = __expf(Ssm[q * 65 + k] - mx);
      Ssm[q * 65 + k] = e;
      sum += e;
    }
    float inv = 1.0f / sum;
    for (int k = 0; k < kv_len; ++k)
      Psm[q * PST + k] = (_Float16)(Ssm[q * 65 + k] * inv);
    for (int k = kv_len; k < 64; ++k)
      Psm[q * PST + k] = (_Float16)0.0f;
  }

  // -------- O = P @ V  (V fragments hoisted, reused 4x) --------
  {
    v16h vf[2][2];
#pragma unroll
    for (int ks = 0; ks < 2; ++ks)
#pragma unroll
      for (int nt = 0; nt < 2; ++nt)
        vf[ks][nt] =
            load16_f16(&Vtsm[(nt * 16 + lm) * PST + ks * 32 + hi * 16]);
#pragma unroll
    for (int mt = 0; mt < 4; ++mt) {
      v8f oacc[2] = {};
#pragma unroll
      for (int ks = 0; ks < 2; ++ks) {
        v16h pa = load_a_f16(&Psm[(mt * 16) * PST + ks * 32], PST, lane);
#pragma unroll
        for (int nt = 0; nt < 2; ++nt)
          oacc[nt] = WMMA_F32_F16(pa, vf[ks][nt], oacc[nt]);
      }
#pragma unroll
      for (int nt = 0; nt < 2; ++nt)
#pragma unroll
        for (int r = 0; r < 8; ++r) {
          int q = mt * 16 + hi * 8 + r;
          O[q * 256 + nt * 16 + lm] = oacc[nt][r];
        }
    }
  }
}

// ---------------------------------------------------------------------------
// st = LayerNorm(st + delta) * w + b   (one wave per 256-wide token)
// ---------------------------------------------------------------------------
__global__ __launch_bounds__(256) void add_ln_kernel(
    float* __restrict__ st, const float* __restrict__ delta,
    const float* __restrict__ w, const float* __restrict__ b, int ntok) {
  int wave = (blockIdx.x * blockDim.x + threadIdx.x) >> 5;
  int lane = threadIdx.x & 31;
  if (wave >= ntok) return;
  float* x0 = st + wave * 256;
  const float* d0 = delta + wave * 256;
  float x[8];
  float s = 0.0f;
#pragma unroll
  for (int i = 0; i < 8; ++i) {
    int c = lane + i * 32;
    x[i] = x0[c] + d0[c];
    s += x[i];
  }
#pragma unroll
  for (int off = 16; off; off >>= 1) s += __shfl_xor(s, off, 32);
  float mu = s * (1.0f / 256.0f);
  float v = 0.0f;
#pragma unroll
  for (int i = 0; i < 8; ++i) {
    float t = x[i] - mu;
    v += t * t;
  }
#pragma unroll
  for (int off = 16; off; off >>= 1) v += __shfl_xor(v, off, 32);
  float rstd = rsqrtf(v * (1.0f / 256.0f) + 1e-5f);
#pragma unroll
  for (int i = 0; i < 8; ++i) {
    int c = lane + i * 32;
    x0[c] = (x[i] - mu) * rstd * w[c] + b[c];
  }
}

// st init: st[b][row][d] = init_part_tokens[row][d] + part_id_emb[row/64][d]
__global__ __launch_bounds__(256) void init_st_kernel(
    const float* __restrict__ itok, const float* __restrict__ pemb,
    float* __restrict__ st) {
  int idx = blockIdx.x * 256 + threadIdx.x;  // grid sized exactly
  int d = idx & 255;
  int row = (idx >> 8) & 511;
  int p = row >> 6;
  st[idx] = itok[row * 256 + d] + pemb[p * 256 + d];
}

// coords: out[tok][0..2] = st[tok] . coord_w[c] + coord_b[c]
__global__ __launch_bounds__(256) void coords_kernel(
    const float* __restrict__ st, const float* __restrict__ cw,
    const float* __restrict__ cb, float* __restrict__ out, int ntok) {
  int wave = (blockIdx.x * blockDim.x + threadIdx.x) >> 5;
  int lane = threadIdx.x & 31;
  if (wave >= ntok) return;
  const float* x = st + wave * 256;
  float s0 = 0, s1 = 0, s2 = 0;
#pragma unroll
  for (int i = 0; i < 8; ++i) {
    int c = lane + i * 32;
    float xv = x[c];
    s0 += xv * cw[c];
    s1 += xv * cw[256 + c];
    s2 += xv * cw[512 + c];
  }
#pragma unroll
  for (int off = 16; off; off >>= 1) {
    s0 += __shfl_xor(s0, off, 32);
    s1 += __shfl_xor(s1, off, 32);
    s2 += __shfl_xor(s2, off, 32);
  }
  if (lane == 0) {
    float* o = out + wave * 3;
    o[0] = s0 + cb[0];
    o[1] = s1 + cb[1];
    o[2] = s2 + cb[2];
  }
}

// ---------------------------------------------------------------------------
extern "C" void kernel_launch(void* const* d_in, const int* in_sizes, int n_in,
                              void* d_out, int out_size, void* d_ws,
                              size_t ws_size, hipStream_t stream) {
  const float* feat        = (const float*)d_in[0];
  const float* img_proj_w  = (const float*)d_in[1];
  const float* img_proj_b  = (const float*)d_in[2];
  const float* init_tokens = (const float*)d_in[3];
  const float* part_id_emb = (const float*)d_in[4];
  const float* local_in_w  = (const float*)d_in[5];
  const float* local_in_b  = (const float*)d_in[6];
  const float* local_out_w = (const float*)d_in[7];
  const float* local_out_b = (const float*)d_in[8];
  const float* cross_in_w  = (const float*)d_in[9];
  const float* cross_in_b  = (const float*)d_in[10];
  const float* cross_out_w = (const float*)d_in[11];
  const float* cross_out_b = (const float*)d_in[12];
  const float* norm1_w     = (const float*)d_in[13];
  const float* norm1_b     = (const float*)d_in[14];
  const float* norm2_w     = (const float*)d_in[15];
  const float* norm2_b     = (const float*)d_in[16];
  const float* coord_w     = (const float*)d_in[17];
  const float* coord_b     = (const float*)d_in[18];

  // Workspace layout (256B aligned)
  char* ws = (char*)d_ws;
  float*    st    = (float*)(ws + 0);            // 32768x256 f32  (33.5 MB)
  float*    img   = (float*)(ws + 33554432);     // 3136x256  f32  ( 3.2 MB)
  _Float16* kvc   = (_Float16*)(ws + 36765696);  // 3136x512  f16  ( 3.2 MB)
  _Float16* qkv   = (_Float16*)(ws + 39976960);  // 32768x768 f16  (50.3 MB; reused for q-proj)
  float*    attnt = (float*)(ws + 90308608);     // 32768x256 f32  (33.5 MB)
  float*    projt = (float*)(ws + 123863040);    // 32768x256 f32  (33.5 MB)

  const float rscale = 0.17677669529663687f;  // 1/sqrt(32)

  // img_tokens = feat @ img_proj_w^T + b    (M=3136, N=256, K=512)
  // 98 mtiles(32) x 4 nstrips = 392 waves -> 49 blocks
  gemm_bias_kernel<float><<<49, 256, 0, stream>>>(
      feat, img_proj_w, img_proj_b, img, 3136, 256, 512);
  // cross K,V (loop invariant): img @ cross_in_w[256:768]^T (M=3136,N=512,K=256)
  // 98 x 8 = 784 waves -> 98 blocks
  gemm_bias_kernel<_Float16><<<98, 256, 0, stream>>>(
      img, cross_in_w + 256 * 256, cross_in_b + 256, kvc, 3136, 512, 256);
  // st init
  init_st_kernel<<<32768, 256, 0, stream>>>(init_tokens, part_id_emb, st);

  for (int i = 0; i < 6; ++i) {
    // local QKV: st @ local_in_w^T (M=32768, N=768, K=256): 1024x12 -> 1536
    gemm_bias_kernel<_Float16><<<1536, 256, 0, stream>>>(
        st, local_in_w, local_in_b, qkv, 32768, 768, 256);
    // local self-attention: 512 groups x 8 heads
    attn_kernel<<<4096, 32, 0, stream>>>(
        qkv, qkv + 256, qkv + 512, attnt,
        /*kv_len=*/64, /*qtiles=*/1, /*q_rs=*/768, /*kv_rs=*/768,
        /*q_gs=*/64 * 768, /*kv_gs=*/64 * 768, /*o_gs=*/64 * 256, rscale);
    // local out proj (M=32768,N=256,K=256): 1024x4 -> 512 blocks
    gemm_bias_kernel<float><<<512, 256, 0, stream>>>(
        attnt, local_out_w, local_out_b, projt, 32768, 256, 256);
    add_ln_kernel<<<4096, 256, 0, stream>>>(
        st, projt, norm1_w + i * 256, norm1_b + i * 256, 32768);

    // cross Q: st @ cross_in_w[0:256]^T
    gemm_bias_kernel<_Float16><<<512, 256, 0, stream>>>(
        st, cross_in_w, cross_in_b, qkv, 32768, 256, 256);
    // cross attention: 64 groups x 8 heads x 8 qtiles
    attn_kernel<<<4096, 32, 0, stream>>>(
        qkv, kvc, kvc + 256, attnt,
        /*kv_len=*/49, /*qtiles=*/8, /*q_rs=*/256, /*kv_rs=*/512,
        /*q_gs=*/512 * 256, /*kv_gs=*/49 * 512, /*o_gs=*/512 * 256, rscale);
    // cross out proj
    gemm_bias_kernel<float><<<512, 256, 0, stream>>>(
        attnt, cross_out_w, cross_out_b, projt, 32768, 256, 256);
    add_ln_kernel<<<4096, 256, 0, stream>>>(
        st, projt, norm2_w + i * 256, norm2_b + i * 256, 32768);
  }

  coords_kernel<<<4096, 256, 0, stream>>>(st, coord_w, coord_b,
                                          (float*)d_out, 32768);
}
